// MultiHeadSelfAttention_78271484002584
// MI455X (gfx1250) — matrix operand order repaired and checked by reading, not compiled, not run on hardware
//
#include <hip/hip_runtime.h>

// ---------------------------------------------------------------------------
// Types
// ---------------------------------------------------------------------------
typedef __attribute__((ext_vector_type(16))) __bf16 v16bf;
typedef __attribute__((ext_vector_type(8)))  __bf16 v8bf;
typedef __attribute__((ext_vector_type(4)))  __bf16 v4bf;
typedef __attribute__((ext_vector_type(8)))  float  v8f;
typedef __attribute__((ext_vector_type(2)))  unsigned long long ull2;
typedef __attribute__((ext_vector_type(4)))  int    v4i;

// ---------------------------------------------------------------------------
// CDNA5 async global->LDS copies (ASYNCcnt path), guarded so the file builds
// on toolchains without the builtin (falls back to load+ds_store staging).
// Builtin signature (from hipcc diagnostic): (AS1 v4i*, AS3 v4i*, imm, imm).
// ---------------------------------------------------------------------------
#if defined(__gfx1250__) &&                                                \
    __has_builtin(__builtin_amdgcn_global_load_async_to_lds_b128) &&       \
    __has_builtin(__builtin_amdgcn_s_wait_asynccnt)
#define HAS_ASYNC_LDS 1
#else
#define HAS_ASYNC_LDS 0
#endif

#if HAS_ASYNC_LDS
typedef __attribute__((address_space(1))) v4i gas_v4i;
typedef __attribute__((address_space(3))) v4i las_v4i;
static __device__ __forceinline__ void async_copy16(const void* g, void* l) {
  // emits global_load_async_to_lds_b128 (16B per lane), tracked by ASYNCcnt
  __builtin_amdgcn_global_load_async_to_lds_b128((gas_v4i*)g, (las_v4i*)l,
                                                 0, 0);
}
static __device__ __forceinline__ void wait_async0() {
  __builtin_amdgcn_s_wait_asynccnt(0);
}
#else
static __device__ __forceinline__ void wait_async0() {}
#endif

// f32 -> bf16 round-to-nearest-even (bit level; no reliance on fp conversion)
static __device__ __forceinline__ __bf16 tobf(float f) {
  union { float f; unsigned u; } x; x.f = f;
  unsigned r = (x.u + 0x7FFFu + ((x.u >> 16) & 1u)) >> 16;
  unsigned short us = (unsigned short)r;
  return __builtin_bit_cast(__bf16, us);
}

// 16-half fragment from 2 aligned 16B LDS/global chunks (contiguous)
static __device__ __forceinline__ v16bf ld16(const __bf16* p) {
  v8bf a = *(const v8bf*)p;
  v8bf b = *(const v8bf*)(p + 8);
  return __builtin_shufflevector(a, b, 0,1,2,3,4,5,6,7,8,9,10,11,12,13,14,15);
}
// A-matrix fragment pattern: halves 0-7 at p, halves 8-15 at p+16 (ISA 7.12.2)
static __device__ __forceinline__ v16bf ld16_gap(const __bf16* p) {
  v8bf a = *(const v8bf*)p;
  v8bf b = *(const v8bf*)(p + 16);
  return __builtin_shufflevector(a, b, 0,1,2,3,4,5,6,7,8,9,10,11,12,13,14,15);
}
static __device__ __forceinline__ v8f wmma_bf16(v16bf a, v16bf b, v8f c) {
  // (neg_a, A, neg_b, B, c_mod, C, reuse_a, reuse_b)
  return __builtin_amdgcn_wmma_f32_16x16x32_bf16(false, a, false, b, (short)0, c,
                                                 false, false);
}

// ---------------------------------------------------------------------------
// Kernel 0: f32 -> bf16 conversion (x and weights), vectorized x4
// ---------------------------------------------------------------------------
__global__ __launch_bounds__(256) void cvt_bf16(const float* __restrict__ s,
                                                __bf16* __restrict__ d, int n4) {
  int i = blockIdx.x * 256 + threadIdx.x;
  if (i < n4) {
    const float4 f = ((const float4*)s)[i];
    v4bf o;
    o[0] = tobf(f.x); o[1] = tobf(f.y); o[2] = tobf(f.z); o[3] = tobf(f.w);
    ((v4bf*)d)[i] = o;
  }
}

// ---------------------------------------------------------------------------
// Kernel 1: tiled WMMA GEMM   C[M,N] = A[M,K] @ W[N,K]^T + bias
//   block tile 128x128, K-step 32, 8 waves, wave tile 32x64 (2x4 WMMA accs).
//   Double-buffered LDS; stage i+1 copied (async global->LDS where available)
//   while stage i feeds the WMMAs.
//   MODE 0: bf16 out row-major | MODE 1: bf16 head-split [b,h,s,hd] | MODE 2: f32
// All M, N multiples of 128 and K of 32 here -> no bounds checks.
// ---------------------------------------------------------------------------
template <int MODE>
__global__ __launch_bounds__(256) void gemm_bf16_kernel(
    const __bf16* __restrict__ A, int lda,
    const __bf16* __restrict__ W,        // [N][K], row stride K
    const float* __restrict__ bias,      // [N]
    void* __restrict__ Cout, int ldc, int K) {
  __shared__ __align__(16) __bf16 As[2][128 * 40];  // pad 8 halves per row
  __shared__ __align__(16) __bf16 Bs[2][128 * 40];

  const int tid = threadIdx.x;
  const int w = tid >> 5, lane = tid & 31;
  const int wm = w & 3, wn = w >> 2;          // 4 waves along M, 2 along N
  const int m = lane & 15, hs = lane >> 4;    // ISA fragment lane decomposition
  const size_t rowBase = (size_t)blockIdx.y * 128;
  const size_t colBase = (size_t)blockIdx.x * 128;

  // stage one 128x32 bf16 tile of A and of W into LDS buffer `buf`
  auto stage = [&](int k0, int buf) {
#pragma unroll
    for (int i = 0; i < 2; ++i) {
      const int chunk = tid * 2 + i;      // 0..511
      const int r = chunk >> 2;           // row 0..127
      const int c = (chunk & 3) << 3;     // half-offset 0,8,16,24
      const __bf16* ga = A + (rowBase + r) * (size_t)lda + k0 + c;
      const __bf16* gw = W + (colBase + r) * (size_t)K + k0 + c;
#if HAS_ASYNC_LDS
      async_copy16(ga, &As[buf][r * 40 + c]);
      async_copy16(gw, &Bs[buf][r * 40 + c]);
#else
      *(ull2*)(&As[buf][r * 40 + c]) = *(const ull2*)ga;
      *(ull2*)(&Bs[buf][r * 40 + c]) = *(const ull2*)gw;
#endif
    }
  };

  v8f acc[2][4];
  for (int i = 0; i < 2; ++i)
    for (int j = 0; j < 4; ++j)
      for (int e = 0; e < 8; ++e) acc[i][j][e] = 0.f;

  stage(0, 0);
  const int nIter = K >> 5;
  for (int it = 0; it < nIter; ++it) {
    const int cur = it & 1;
    wait_async0();        // my stage-`it` copies have landed in LDS
    __syncthreads();      // everyone's stage-`it` landed; buf cur^1 free again
    if (it + 1 < nIter) stage((it + 1) << 5, cur ^ 1);  // overlaps compute

    v16bf af[2], bfg[4];
#pragma unroll
    for (int mi = 0; mi < 2; ++mi)
      af[mi] = ld16_gap(&As[cur][(wm * 32 + mi * 16 + m) * 40 + hs * 8]);
#pragma unroll
    for (int ni = 0; ni < 4; ++ni)
      bfg[ni] = ld16_gap(&Bs[cur][(wn * 64 + ni * 16 + m) * 40 + hs * 8]);
#pragma unroll
    for (int mi = 0; mi < 2; ++mi)
#pragma unroll
      for (int ni = 0; ni < 4; ++ni)
        acc[mi][ni] = wmma_bf16(af[mi], bfg[ni], acc[mi][ni]);
  }

  // epilogue: C layout lane mapping  row = r + 8*hs, col = m  (per 16x16 tile)
#pragma unroll
  for (int mi = 0; mi < 2; ++mi)
#pragma unroll
    for (int ni = 0; ni < 4; ++ni) {
      const size_t col = colBase + wn * 64 + ni * 16 + m;
      const float bv = bias[col];
#pragma unroll
      for (int r = 0; r < 8; ++r) {
        const size_t row = rowBase + wm * 32 + mi * 16 + r + 8 * hs;
        const float val = acc[mi][ni][r] + bv;
        if constexpr (MODE == 0) {
          ((__bf16*)Cout)[row * (size_t)ldc + col] = tobf(val);
        } else if constexpr (MODE == 2) {
          ((float*)Cout)[row * (size_t)ldc + col] = val;
        } else {  // head split: row = b*2048+s, col = h*128+hd
          const size_t bb = row >> 11, s = row & 2047;
          const size_t hh = col >> 7, hd = col & 127;
          ((__bf16*)Cout)[((bb * 8 + hh) * 2048 + s) * 128 + hd] = tobf(val);
        }
      }
    }
}

// ---------------------------------------------------------------------------
// Kernel 2: fused flash attention.
//   grid = (S/128 q-blocks, B*H). 256 threads = 8 waves; wave owns 16 q rows.
//   Q stripe resident in registers as 4 A-fragments (HD=128 = 4 x K32).
//   K staged to LDS [key][hd] via async global->LDS; V staged transposed
//   [hd][key] (through VGPRs, it needs the transpose) so PV B-fragments are
//   contiguous LDS reads. Online softmax via shfl_xor reductions over the
//   16-lane halves of a wave32.
// ---------------------------------------------------------------------------
__global__ __launch_bounds__(256) void flash_attn(
    const __bf16* __restrict__ Qh, const __bf16* __restrict__ Kh,
    const __bf16* __restrict__ Vh, const int* __restrict__ amask,
    __bf16* __restrict__ Ob) {
  constexpr int S = 2048, HD = 128, D = 1024;
  const float SCALE = 0.08838834764831845f;  // 1/sqrt(128)

  __shared__ __align__(16) __bf16 Kt[64 * 136];     // [key][hd], pad 8
  __shared__ __align__(16) __bf16 Vt[128 * 72];     // [hd][key], pad 8
  __shared__ __align__(16) __bf16 Ps[8 * 16 * 40];  // per-wave P scratch

  const int tid = threadIdx.x, w = tid >> 5, lane = tid & 31;
  const int m = lane & 15, hs = lane >> 4;
  const int bh = blockIdx.y;
  const int b = bh >> 3, h = bh & 7;
  const int qtile = blockIdx.x * 128 + w * 16;

  const __bf16* Qp = Qh + (size_t)bh * S * HD;
  const __bf16* Kp = Kh + (size_t)bh * S * HD;
  const __bf16* Vp = Vh + (size_t)bh * S * HD;

  // resident Q fragments (A-matrix layout), read straight from global
  v16bf qf[4];
#pragma unroll
  for (int f = 0; f < 4; ++f)
    qf[f] = ld16_gap(Qp + (size_t)(qtile + m) * HD + f * 32 + hs * 8);

  float mrow[8], lrow[8];
  v8f oacc[8];
#pragma unroll
  for (int r = 0; r < 8; ++r) { mrow[r] = -1e30f; lrow[r] = 0.f; }
#pragma unroll
  for (int t = 0; t < 8; ++t)
    for (int e = 0; e < 8; ++e) oacc[t][e] = 0.f;

  __bf16* Pw = Ps + w * 16 * 40;

  for (int kb = 0; kb < S; kb += 64) {
    __syncthreads();  // everyone done reading Kt/Vt from previous block
    // stage K (async, row-major) and V (transposed via VGPRs) for 64 keys
#pragma unroll
    for (int i = 0; i < 4; ++i) {
      const int chunk = tid + i * 256;      // 0..1023
      const int key = chunk >> 4;           // 0..63
      const int c = (chunk & 15) << 3;      // hd half-offset 0..120
      const __bf16* gk = Kp + (size_t)(kb + key) * HD + c;
#if HAS_ASYNC_LDS
      async_copy16(gk, Kt + key * 136 + c);
#else
      *(ull2*)(Kt + key * 136 + c) = *(const ull2*)gk;
#endif
      v8bf vv = *(const v8bf*)(Vp + (size_t)(kb + key) * HD + c);
#pragma unroll
      for (int j = 0; j < 8; ++j) Vt[(c + j) * 72 + key] = vv[j];
    }
    wait_async0();
    __syncthreads();

#pragma unroll
    for (int c32 = 0; c32 < 64; c32 += 32) {
      // scores for keys [kb+c32, kb+c32+32): two 16x16 tiles
      v8f s0, s1;
      for (int e = 0; e < 8; ++e) { s0[e] = 0.f; s1[e] = 0.f; }
#pragma unroll
      for (int f = 0; f < 4; ++f) {
        v16bf b0 = ld16_gap(Kt + (c32 + m) * 136 + f * 32 + hs * 8);
        v16bf b1 = ld16_gap(Kt + (c32 + 16 + m) * 136 + f * 32 + hs * 8);
        s0 = wmma_bf16(qf[f], b0, s0);
        s1 = wmma_bf16(qf[f], b1, s1);
      }

      // masked online softmax, per C-tile row r (row = r + 8*hs)
      const int keyg0 = kb + c32 + m;
#pragma unroll
      for (int r = 0; r < 8; ++r) {
        const int qrow = qtile + r + 8 * hs;
        const int* mp = amask + (size_t)qrow * S;
        float v0 = mp[keyg0]      ? -1e30f : s0[r] * SCALE;
        float v1 = mp[keyg0 + 16] ? -1e30f : s1[r] * SCALE;
        float cmax = fmaxf(v0, v1);
#pragma unroll
        for (int off = 1; off < 16; off <<= 1)
          cmax = fmaxf(cmax, __shfl_xor(cmax, off, 32));
        const float mnew = fmaxf(mrow[r], cmax);
        const float alpha = __expf(mrow[r] - mnew);
        const float p0 = __expf(v0 - mnew);
        const float p1 = __expf(v1 - mnew);
        float rs = p0 + p1;
#pragma unroll
        for (int off = 1; off < 16; off <<= 1) rs += __shfl_xor(rs, off, 32);
        lrow[r] = lrow[r] * alpha + rs;
        mrow[r] = mnew;
#pragma unroll
        for (int t = 0; t < 8; ++t) oacc[t][r] *= alpha;
        const int prow = r + 8 * hs;
        Pw[prow * 40 + m]      = tobf(p0);
        Pw[prow * 40 + m + 16] = tobf(p1);
      }

      // P (16x32) as A-fragment, then PV across all 8 hd tiles
      v16bf pf = ld16_gap(Pw + m * 40 + hs * 8);
#pragma unroll
      for (int t = 0; t < 8; ++t) {
        v16bf bv = ld16_gap(Vt + (t * 16 + m) * 72 + c32 + hs * 8);
        oacc[t] = wmma_bf16(pf, bv, oacc[t]);
      }
    }
  }

  // epilogue: O[b, s, h*HD + hd] in bf16 for the final projection
#pragma unroll
  for (int t = 0; t < 8; ++t) {
    const int hd = t * 16 + m;
#pragma unroll
    for (int r = 0; r < 8; ++r) {
      const int qrow = qtile + r + 8 * hs;
      const float val = oacc[t][r] / lrow[r];
      Ob[((size_t)b * S + qrow) * D + h * HD + hd] = tobf(val);
    }
  }
}

// ---------------------------------------------------------------------------
// Host orchestration
// ---------------------------------------------------------------------------
extern "C" void kernel_launch(void* const* d_in, const int* in_sizes, int n_in,
                              void* d_out, int out_size, void* d_ws,
                              size_t ws_size, hipStream_t stream) {
  (void)in_sizes; (void)n_in; (void)out_size; (void)ws_size;

  const float* x     = (const float*)d_in[0];
  const int*   amask = (const int*)d_in[1];
  const float* qkv_w = (const float*)d_in[2];
  const float* qkv_b = (const float*)d_in[3];
  const float* q_w   = (const float*)d_in[4];
  const float* q_b   = (const float*)d_in[5];
  const float* k_w   = (const float*)d_in[6];
  const float* k_b   = (const float*)d_in[7];
  const float* v_w   = (const float*)d_in[8];
  const float* v_b   = (const float*)d_in[9];
  const float* out_w = (const float*)d_in[10];
  const float* out_b = (const float*)d_in[11];
  float* out = (float*)d_out;

  constexpr size_t M = 8192;  // B*S
  __bf16* p = (__bf16*)d_ws;
  __bf16* x_bf    = p; p += M * 1024;
  __bf16* qkvw_bf = p; p += (size_t)3072 * 1024;
  __bf16* qw_bf   = p; p += (size_t)1024 * 1024;
  __bf16* kw_bf   = p; p += (size_t)1024 * 1024;
  __bf16* vw_bf   = p; p += (size_t)1024 * 1024;
  __bf16* ow_bf   = p; p += (size_t)1024 * 1024;
  __bf16* qkv_bf  = p; p += M * 3072;
  __bf16* Qh      = p; p += M * 1024;
  __bf16* Kh      = p; p += M * 1024;
  __bf16* Vh      = p; p += M * 1024;
  __bf16* Ob      = x_bf;  // x_bf dead after the QKV GEMM; reuse for O

  auto cvt = [&](const float* s, __bf16* d, size_t n) {
    const int n4 = (int)(n / 4);
    cvt_bf16<<<dim3((n4 + 255) / 256), dim3(256), 0, stream>>>(s, d, n4);
  };
  cvt(x, x_bf, M * 1024);
  cvt(qkv_w, qkvw_bf, (size_t)3072 * 1024);
  cvt(q_w, qw_bf, (size_t)1024 * 1024);
  cvt(k_w, kw_bf, (size_t)1024 * 1024);
  cvt(v_w, vw_bf, (size_t)1024 * 1024);
  cvt(out_w, ow_bf, (size_t)1024 * 1024);

  // qkv = x @ qkv_w^T + qkv_b          [8192, 3072] bf16
  gemm_bf16_kernel<0><<<dim3(24, 64), dim3(256), 0, stream>>>(
      x_bf, 1024, qkvw_bf, qkv_b, qkv_bf, 3072, 1024);

  // per-stream head projections -> [B, H, S, HD] bf16
  gemm_bf16_kernel<1><<<dim3(8, 64), dim3(256), 0, stream>>>(
      qkv_bf + 0,    3072, qw_bf, q_b, Qh, 0, 1024);
  gemm_bf16_kernel<1><<<dim3(8, 64), dim3(256), 0, stream>>>(
      qkv_bf + 1024, 3072, kw_bf, k_b, Kh, 0, 1024);
  gemm_bf16_kernel<1><<<dim3(8, 64), dim3(256), 0, stream>>>(
      qkv_bf + 2048, 3072, vw_bf, v_b, Vh, 0, 1024);

  // fused masked-softmax attention -> O [B, S, D] bf16
  flash_attn<<<dim3(16, 32), dim3(256), 0, stream>>>(Qh, Kh, Vh, amask, Ob);

  // final projection, f32 output
  gemm_bf16_kernel<2><<<dim3(8, 64), dim3(256), 0, stream>>>(
      Ob, 1024, ow_bf, out_b, out, 1024, 1024);
}
